// TopologicalCoordinateGenerator_22024592294441
// MI455X (gfx1250) — matrix operand order repaired and checked by reading, not compiled, not run on hardware
//
#include <hip/hip_runtime.h>
#include <hip/hip_bf16.h>
#include <math.h>

// ---------------------------------------------------------------------------
// Types for WMMA (gfx1250, wave32)
// ---------------------------------------------------------------------------
typedef __attribute__((ext_vector_type(16))) _Float16 v16h;
typedef __attribute__((ext_vector_type(8)))  _Float16 v8h;
typedef __attribute__((ext_vector_type(8)))  float    v8f;

#define PI_F 3.14159265358979323846f

__device__ __forceinline__ float gelu_f(float x) {
    float x3 = x * x * x;
    return 0.5f * x * (1.0f + tanhf(0.7978845608028654f * (x + 0.044715f * x3)));
}

__device__ __forceinline__ v8f wmma_f16(v16h a, v16h b, v8f c) {
    return __builtin_amdgcn_wmma_f32_16x16x32_f16(
        /*neg_a=*/false, a, /*neg_b=*/false, b,
        /*c_mod=*/(short)0, c, /*reuse_a=*/false, /*reuse_b=*/false);
}

// ---------------------------------------------------------------------------
// Fragment layouts (CDNA5 ISA 7.12.2), 16x16x32 f16:
//   A 16x32: lanes 0-15 row m0+l, elems = 16 contiguous K k0..k0+15;
//            lanes 16-31 same rows, +16 on K (this code's order, not the ISA table order).
//   B 32x16: lane -> column, elems = 16 contiguous K (halves split at +16), same order as A.
//   C/D: vgpr r -> row m0 + r + 8*(lane>=16), col n0 + (lane&15).
// ---------------------------------------------------------------------------

// ---------------------------------------------------------------------------
// Blocked WMMA GEMM: block = 8 waves, each wave a 32x64 C tile (2x4 subtiles,
// 8 accumulators); block covers 256 rows x 64 cols. B panel (64 x 32 f16 per
// K-step) is staged through double-buffered LDS and shared by all 8 waves
// (8x less B traffic); A fragments are reused across the 4 N-subtiles.
// Requires: M % 256 == 0, Npad % 64 == 0, Kpad % 32 == 0. No divergence
// before the WMMAs -> EXEC all-ones as required.
// ---------------------------------------------------------------------------
__global__ void __launch_bounds__(256)
k_gemm_blk(const _Float16* __restrict__ A, int lda,
           const _Float16* __restrict__ Wt,           // [Npad, Kpad] f16, zero-padded
           const float* __restrict__ bias,
           _Float16* __restrict__ Out, int ldo,
           int Ntrue, int Kpad, int nblocks_n, int act)
{
    __shared__ _Float16 lb[2 * 64 * 32];              // double-buffered B panel, 8 KB

    const int wave = threadIdx.x >> 5;
    const int lane = threadIdx.x & 31;
    const int l16  = lane & 15;
    const int half = lane >> 4;

    int bm = blockIdx.x / nblocks_n;
    int bn = blockIdx.x - bm * nblocks_n;
    int m0 = bm * 256 + wave * 32;
    int n0 = bn * 64;

    const _Float16* Ar0 = A + (size_t)(m0 + l16) * lda + half * 16;
    const _Float16* Ar1 = Ar0 + (size_t)16 * lda;

    // B staging: thread t copies 8 halfs of LDS row (t>>2), segment (t&3)
    const int tr   = threadIdx.x >> 2;                // 0..63  (panel row)
    const int tseg = threadIdx.x & 3;                 // 0..3   (8-half segment)
    const _Float16* Bsrc = Wt + (size_t)(n0 + tr) * Kpad + tseg * 8;
    _Float16* Bd0 = lb + tr * 32 + tseg * 8;

    v8f c[2][4] = {};

    // prologue: stage first K chunk into buffer 0
    *(v8h*)Bd0 = *(const v8h*)Bsrc;

    int p = 0;
    for (int k0 = 0; k0 < Kpad; k0 += 32) {
        __syncthreads();                              // buf p ready / prev reads done
        if (k0 + 32 < Kpad) {
            __builtin_prefetch(Ar0 + k0 + 32, 0, 3);  // gfx1250 global_prefetch path
            __builtin_prefetch(Ar1 + k0 + 32, 0, 3);
            *(v8h*)(Bd0 + (p ^ 1) * 2048) = *(const v8h*)(Bsrc + k0 + 32);
        }
        union { v16h v; v8h h[2]; } a0, a1;
        a0.h[0] = *(const v8h*)(Ar0 + k0);
        a0.h[1] = *(const v8h*)(Ar0 + k0 + 8);
        a1.h[0] = *(const v8h*)(Ar1 + k0);
        a1.h[1] = *(const v8h*)(Ar1 + k0 + 8);
        const _Float16* lbp = lb + p * 2048;
#pragma unroll
        for (int j = 0; j < 4; ++j) {
            v16h b = *(const v16h*)(lbp + (j * 16 + l16) * 32 + half * 16);
            c[0][j] = wmma_f16(a0.v, b, c[0][j]);
            c[1][j] = wmma_f16(a1.v, b, c[1][j]);
        }
        p ^= 1;
    }

#pragma unroll
    for (int i = 0; i < 2; ++i) {
#pragma unroll
        for (int j = 0; j < 4; ++j) {
            int col = n0 + j * 16 + l16;
            if (col < Ntrue) {
                float bv = bias ? bias[col] : 0.0f;
#pragma unroll
                for (int r = 0; r < 8; ++r) {
                    int row = m0 + i * 16 + r + half * 8;
                    float v = c[i][j][r] + bv;
                    if (act == 1)      v = gelu_f(v);
                    else if (act == 2) v = tanhf(v);
                    Out[(size_t)row * ldo + col] = (_Float16)v;
                }
            }
        }
    }
}

// ---------------------------------------------------------------------------
// Simple WMMA GEMM (one 16x16 tile per wave) for the small-N layers
// (ds0 N=32, heads-out N=3, mlp-out N=3).
// ---------------------------------------------------------------------------
__global__ void __launch_bounds__(256)
k_gemm(const _Float16* __restrict__ A, int lda,
       const _Float16* __restrict__ Wt,
       const float* __restrict__ bias,
       _Float16* __restrict__ Out, int ldo,
       int M, int Ntrue, int Kpad, int tiles_n, int act)
{
    const int wave = threadIdx.x >> 5;
    const int lane = threadIdx.x & 31;
    int tile = blockIdx.x * 8 + wave;
    int tiles_m = M >> 4;
    if (tile >= tiles_m * tiles_n) return;            // wave-uniform: EXEC stays all-ones
    int tm = tile / tiles_n;
    int tn = tile - tm * tiles_n;
    int m0 = tm << 4, n0 = tn << 4;

    const int l16  = lane & 15;
    const int half = lane >> 4;

    const _Float16* Arow = A  + (size_t)(m0 + l16) * lda  + half * 16;
    const _Float16* Brow = Wt + (size_t)(n0 + l16) * Kpad + half * 16;

    v8f c = {};
    for (int k0 = 0; k0 < Kpad; k0 += 32) {
        if (k0 + 32 < Kpad) {
            __builtin_prefetch(Arow + k0 + 32, 0, 3);
            __builtin_prefetch(Brow + k0 + 32, 0, 3);
        }
        union { v16h v; v8h h[2]; } a;
        a.h[0] = *(const v8h*)(Arow + k0);
        a.h[1] = *(const v8h*)(Arow + k0 + 8);
        v16h b = *(const v16h*)(Brow + k0);
        c = wmma_f16(a.v, b, c);
    }

    int col = n0 + l16;
    if (col < Ntrue) {
        float bv = bias ? bias[col] : 0.0f;
#pragma unroll
        for (int r = 0; r < 8; ++r) {
            int row = m0 + r + half * 8;
            float v = c[r] + bv;
            if (act == 1)      v = gelu_f(v);
            else if (act == 2) v = tanhf(v);
            Out[(size_t)row * ldo + col] = (_Float16)v;
        }
    }
}

// ---------------------------------------------------------------------------
// Weight prep: W f32 [Ktrue, Ntrue] (row-major, HWIO convs flatten to this)
//   -> Wt f16 [Npad, Kpad], transposed + zero padded.
// ---------------------------------------------------------------------------
__global__ void k_prep_w(const float* __restrict__ W, _Float16* __restrict__ Wt,
                         int Ktrue, int Ntrue, int Kpad, int total)
{
    int i = blockIdx.x * blockDim.x + threadIdx.x;
    if (i >= total) return;
    int k = i % Kpad, n = i / Kpad;
    float v = 0.0f;
    if (k < Ktrue && n < Ntrue) v = W[(size_t)k * Ntrue + n];
    Wt[i] = (_Float16)v;
}

// ---------------------------------------------------------------------------
// RGB [-1,1] -> YCbCr (scaled), f32 -> f16
// ---------------------------------------------------------------------------
__global__ void k_ycbcr(const float* __restrict__ img, _Float16* __restrict__ out, int npix)
{
    int i = blockIdx.x * blockDim.x + threadIdx.x;
    if (i >= npix) return;
    float r = (img[(size_t)i * 3 + 0] + 1.0f) * 0.5f;
    float g = (img[(size_t)i * 3 + 1] + 1.0f) * 0.5f;
    float b = (img[(size_t)i * 3 + 2] + 1.0f) * 0.5f;
    float y  =  0.299f    * r + 0.587f    * g + 0.114f    * b;
    float cb = -0.168736f * r - 0.331264f * g + 0.5f      * b;
    float cr =  0.5f      * r - 0.418688f * g - 0.081312f * b;
    out[(size_t)i * 3 + 0] = (_Float16)(y * 2.0f - 1.0f);
    out[(size_t)i * 3 + 1] = (_Float16)(cb * 2.0f);
    out[(size_t)i * 3 + 2] = (_Float16)(cr * 2.0f);
}

// ---------------------------------------------------------------------------
// im2col (NHWC, SAME padding), f16 -> f16 patch matrix [B*Ho*Ho, Kpad]
// k index = (kh*ks + kw)*Cin + ci  (matches HWIO weight flattening)
// ---------------------------------------------------------------------------
__global__ void k_im2col(const _Float16* __restrict__ src, _Float16* __restrict__ dst,
                         int Hin, int Cin, int Hout, int ks, int stride, int pad,
                         int Ktrue, int Kpad, long long total)
{
    long long i = (long long)blockIdx.x * blockDim.x + threadIdx.x;
    if (i >= total) return;
    int k = (int)(i % Kpad);
    long long row = i / Kpad;
    int ow = (int)(row % Hout); long long t = row / Hout;
    int oh = (int)(t % Hout);
    int b  = (int)(t / Hout);
    _Float16 v = (_Float16)0.0f;
    if (k < Ktrue) {
        int kc = ks * Cin;
        int kh = k / kc, rem = k - kh * kc;
        int kw = rem / Cin, ci = rem - kw * Cin;
        int ih = oh * stride - pad + kh;
        int iw = ow * stride - pad + kw;
        if (ih >= 0 && ih < Hin && iw >= 0 && iw < Hin)
            v = src[(((size_t)b * Hin + ih) * Hin + iw) * Cin + ci];
    }
    dst[i] = v;
}

// ---------------------------------------------------------------------------
// Path statistics: per (b,pixel) row, 3 channels x 16 path steps.
// ---------------------------------------------------------------------------
__global__ void k_stats(const _Float16* __restrict__ hr0, const _Float16* __restrict__ hr1,
                        const _Float16* __restrict__ hr2, _Float16* __restrict__ stats, int M)
{
    int row = blockIdx.x * blockDim.x + threadIdx.x;
    if (row >= M) return;
    const _Float16* hrs[3] = {hr0, hr1, hr2};
    for (int c = 0; c < 3; ++c) {
        const _Float16* hr = hrs[c];
        float r0 = (float)hr[(size_t)row * 16 + 0];
        float r1 = (float)hr[(size_t)row * 16 + 1];
        float r2 = (float)hr[(size_t)row * 16 + 2];
        float delta  = tanhf(r0) * PI_F;
        float chi    = tanhf(r1) * (PI_F * 0.25f);
        float radius = (1.0f / (1.0f + expf(-r2))) * (PI_F * 0.5f);
        float sp = 0.f, sa = 0.f, sa2 = 0.f, pmin = 1e30f, pmax = -1e30f;
        for (int s = 0; s < 16; ++s) {
            float th = (2.0f * PI_F) * (float)s / 15.0f;   // linspace(0, 2pi, 16)
            float dp = delta + radius * cosf(th);
            float cp = chi   + radius * sinf(th);
            float re = cosf(dp * 0.5f) + 1e-8f;
            float im = sinf(dp * 0.5f) * sinf(2.0f * cp);
            float ph = atan2f(im, re);
            float amp = sqrtf(re * re + im * im);
            sp += ph; sa += amp; sa2 += amp * amp;
            pmin = fminf(pmin, ph); pmax = fmaxf(pmax, ph);
        }
        float mp = sp * (1.0f / 16.0f), ma = sa * (1.0f / 16.0f);
        float var = sa2 * (1.0f / 16.0f) - ma * ma;
        float sd = sqrtf(fmaxf(0.0f, var));
        size_t o = (size_t)row * 32 + c * 5;
        stats[o + 0] = (_Float16)mp;
        stats[o + 1] = (_Float16)(pmax - pmin);
        stats[o + 2] = (_Float16)ma;
        stats[o + 3] = (_Float16)sd;
        stats[o + 4] = (_Float16)radius;
    }
    for (int k = 15; k < 32; ++k) stats[(size_t)row * 32 + k] = (_Float16)0.0f;
}

// ---------------------------------------------------------------------------
// Positional encoding into hcat[:, 256..297] (broadcast over B) and zero-pad
// hcat[:, 1066..1087] (K padding of MLP layer 0).
// ---------------------------------------------------------------------------
__global__ void k_posenc(const float* __restrict__ coords, _Float16* __restrict__ hcat, int N)
{
    int n = blockIdx.x * blockDim.x + threadIdx.x;
    if (n >= N) return;
    float e[42];
    float cx = coords[(size_t)n * 2 + 0], cy = coords[(size_t)n * 2 + 1];
    e[0] = cx; e[1] = cy;
    for (int f = 0; f < 10; ++f) {
        float fr = (float)(1 << f) * PI_F;
        e[2 + f * 4 + 0] = sinf(cx * fr);
        e[2 + f * 4 + 1] = sinf(cy * fr);
        e[2 + f * 4 + 2] = cosf(cx * fr);
        e[2 + f * 4 + 3] = cosf(cy * fr);
    }
    for (int b = 0; b < 4; ++b) {
        size_t base = ((size_t)b * N + n) * 1088 + 256;
        for (int j = 0; j < 42; ++j)    hcat[base + j] = (_Float16)e[j];
        for (int j = 810; j < 832; ++j) hcat[base + j] = (_Float16)0.0f;
    }
}

// ---------------------------------------------------------------------------
// Bilinear downsample (half-pixel centers) [B,Hs,Hs,256] -> [B,Hd,Hd,256]
// ---------------------------------------------------------------------------
__global__ void k_resize(const _Float16* __restrict__ src, _Float16* __restrict__ dst,
                         int Hs, int Hd, long long total)
{
    long long i = (long long)blockIdx.x * blockDim.x + threadIdx.x;
    if (i >= total) return;
    int c = (int)(i % 256); long long t = i / 256;
    int x = (int)(t % Hd); t /= Hd;
    int y = (int)(t % Hd);
    int b = (int)(t / Hd);
    float scale = (float)Hs / (float)Hd;
    float sy = fminf(fmaxf((y + 0.5f) * scale - 0.5f, 0.0f), (float)(Hs - 1));
    float sx = fminf(fmaxf((x + 0.5f) * scale - 0.5f, 0.0f), (float)(Hs - 1));
    int y0 = min((int)sy, Hs - 2);
    int x0 = min((int)sx, Hs - 2);
    float fy = sy - y0, fx = sx - x0;
    const _Float16* p = src + (((size_t)b * Hs + y0) * Hs + x0) * 256 + c;
    float v00 = (float)p[0],                 v01 = (float)p[256];
    float v10 = (float)p[(size_t)Hs * 256],  v11 = (float)p[(size_t)Hs * 256 + 256];
    float v = (1 - fy) * ((1 - fx) * v00 + fx * v01) + fy * ((1 - fx) * v10 + fx * v11);
    dst[i] = (_Float16)v;
}

// ---------------------------------------------------------------------------
// Bilinear sampling of one pyramid level at N coords -> hcat cols
// ---------------------------------------------------------------------------
__global__ void k_sample(const _Float16* __restrict__ g, int Hg,
                         const float* __restrict__ coords,
                         _Float16* __restrict__ hcat, int coloff, int N)
{
    int t = blockIdx.x * blockDim.x + threadIdx.x;
    if (t >= 4 * N) return;
    int b = t / N, n = t - b * N;
    float rr = fminf(fmaxf((coords[(size_t)n * 2 + 0] + 1.0f) * 0.5f * (Hg - 1), 0.0f), (float)(Hg - 1));
    float cc = fminf(fmaxf((coords[(size_t)n * 2 + 1] + 1.0f) * 0.5f * (Hg - 1), 0.0f), (float)(Hg - 1));
    int r0 = min((int)rr, Hg - 2);
    int c0 = min((int)cc, Hg - 2);
    float fr = rr - r0, fc = cc - c0;
    float w00 = (1 - fr) * (1 - fc), w01 = (1 - fr) * fc;
    float w10 = fr * (1 - fc),       w11 = fr * fc;
    const _Float16* p00 = g + (((size_t)b * Hg + r0) * Hg + c0) * 256;
    const _Float16* p10 = p00 + (size_t)Hg * 256;
    size_t ob = ((size_t)b * N + n) * 1088 + coloff;
    for (int c = 0; c < 256; ++c) {
        float v = w00 * (float)p00[c] + w01 * (float)p00[256 + c] +
                  w10 * (float)p10[c] + w11 * (float)p10[256 + c];
        hcat[ob + c] = (_Float16)v;
    }
}

// ---------------------------------------------------------------------------
// Final: raw (already tanh'ed) YCbCr -> RGB, clip, write f32 output
// ---------------------------------------------------------------------------
__global__ void k_final(const _Float16* __restrict__ raw, float* __restrict__ out, int M)
{
    int i = blockIdx.x * blockDim.x + threadIdx.x;
    if (i >= M) return;
    float y  = ((float)raw[(size_t)i * 16 + 0] + 1.0f) * 0.5f;
    float cb = (float)raw[(size_t)i * 16 + 1] * 0.5f;
    float cr = (float)raw[(size_t)i * 16 + 2] * 0.5f;
    float r = y + 1.402f * cr;
    float g = y - 0.344136f * cb - 0.714136f * cr;
    float b = y + 1.772f * cb;
    out[(size_t)i * 3 + 0] = fminf(fmaxf(r, 0.0f), 1.0f) * 2.0f - 1.0f;
    out[(size_t)i * 3 + 1] = fminf(fmaxf(g, 0.0f), 1.0f) * 2.0f - 1.0f;
    out[(size_t)i * 3 + 2] = fminf(fmaxf(b, 0.0f), 1.0f) * 2.0f - 1.0f;
}

// ---------------------------------------------------------------------------
// Host orchestration
// ---------------------------------------------------------------------------
static inline void launch_gemm(hipStream_t s, const _Float16* A, int lda,
                               const _Float16* Wt, const float* bias,
                               _Float16* Out, int ldo,
                               int M, int Ntrue, int Kpad, int Npad, int act)
{
    if ((M % 256) == 0 && (Npad % 64) == 0) {
        int nbn = Npad / 64;
        int blocks = (M / 256) * nbn;
        k_gemm_blk<<<blocks, 256, 0, s>>>(A, lda, Wt, bias, Out, ldo, Ntrue, Kpad, nbn, act);
    } else {
        int tiles = (M / 16) * (Npad / 16);
        int blocks = (tiles + 7) / 8;
        k_gemm<<<blocks, 256, 0, s>>>(A, lda, Wt, bias, Out, ldo, M, Ntrue, Kpad, Npad / 16, act);
    }
}

static inline void launch_prep(hipStream_t s, const float* W, _Float16* Wt,
                               int Kt, int Nt, int Kp, int Np)
{
    int total = Np * Kp;
    k_prep_w<<<(total + 255) / 256, 256, 0, s>>>(W, Wt, Kt, Nt, Kp, total);
}

extern "C" void kernel_launch(void* const* d_in, const int* in_sizes, int n_in,
                              void* d_out, int out_size, void* d_ws, size_t ws_size,
                              hipStream_t stream)
{
    const int Bn = 4, N = 16384, BN = 65536;

    const float* img    = (const float*)d_in[0];
    const float* coords = (const float*)d_in[1];
    const float* ds_w[3]  = {(const float*)d_in[2],  (const float*)d_in[4],  (const float*)d_in[6]};
    const float* ds_b[3]  = {(const float*)d_in[3],  (const float*)d_in[5],  (const float*)d_in[7]};
    const float* fin_w    =  (const float*)d_in[8];
    const float* fin_b    =  (const float*)d_in[9];
    const float* c1_w[3]  = {(const float*)d_in[10], (const float*)d_in[14], (const float*)d_in[18]};
    const float* c1_b[3]  = {(const float*)d_in[11], (const float*)d_in[15], (const float*)d_in[19]};
    const float* ho_w[3]  = {(const float*)d_in[12], (const float*)d_in[16], (const float*)d_in[20]};
    const float* ho_b[3]  = {(const float*)d_in[13], (const float*)d_in[17], (const float*)d_in[21]};
    const float* proj_w   =  (const float*)d_in[22];
    const float* proj_b   =  (const float*)d_in[23];
    const float* mlp_w[4] = {(const float*)d_in[24], (const float*)d_in[26], (const float*)d_in[28], (const float*)d_in[30]};
    const float* mlp_b[4] = {(const float*)d_in[25], (const float*)d_in[27], (const float*)d_in[29], (const float*)d_in[31]};
    const float* mo_w     =  (const float*)d_in[32];
    const float* mo_b     =  (const float*)d_in[33];

    // ---- workspace bump allocator -----------------------------------------
    char* ws = (char*)d_ws;
    size_t off = 0;
    auto alloc = [&](size_t bytes) -> char* {
        char* p = ws + off;
        off = (off + bytes + 255) & ~(size_t)255;
        return p;
    };

    // persistent: transposed f16 weights [Npad, Kpad]
    _Float16* wt_ds[3]; _Float16* wt_c1[3]; _Float16* wt_ho[3];
    wt_ds[0] = (_Float16*)alloc((size_t)32  * 64   * 2);
    wt_ds[1] = (_Float16*)alloc((size_t)64  * 512  * 2);
    wt_ds[2] = (_Float16*)alloc((size_t)128 * 1024 * 2);
    _Float16* wt_fin = (_Float16*)alloc((size_t)256 * 1152 * 2);
    for (int c = 0; c < 3; ++c) wt_c1[c] = (_Float16*)alloc((size_t)128 * 256 * 2);
    for (int c = 0; c < 3; ++c) wt_ho[c] = (_Float16*)alloc((size_t)16  * 128 * 2);
    _Float16* wt_proj = (_Float16*)alloc((size_t)256 * 32   * 2);
    _Float16* wt_m0   = (_Float16*)alloc((size_t)256 * 832  * 2);
    _Float16* wt_m1   = (_Float16*)alloc((size_t)256 * 256  * 2);
    _Float16* wt_m2   = (_Float16*)alloc((size_t)256 * 256  * 2);
    _Float16* wt_m3   = (_Float16*)alloc((size_t)256 * 1088 * 2);
    _Float16* wt_mo   = (_Float16*)alloc((size_t)16  * 256  * 2);

    // persistent: feature pyramid + MLP ping-pong
    _Float16* feat = (_Float16*)alloc((size_t)16384 * 256 * 2);           // [4,64,64,256]
    _Float16* pyr1 = (_Float16*)alloc((size_t)4 * 32 * 32 * 256 * 2);
    _Float16* pyr2 = (_Float16*)alloc((size_t)4 * 16 * 16 * 256 * 2);
    _Float16* h1   = (_Float16*)alloc((size_t)BN * 256 * 2);
    _Float16* h2   = (_Float16*)alloc((size_t)BN * 256 * 2);
    _Float16* raw  = (_Float16*)alloc((size_t)BN * 16 * 2);

    // hcat [BN, 1088]: cols 0..255 = MLP layer-2 out, 256..297 = posenc,
    // 298..1065 = sampled pyramid, 1066..1087 = zero pad. Conv-phase scratch
    // is aliased inside this region (disjoint lifetimes).
    _Float16* hcat = (_Float16*)alloc((size_t)BN * 1088 * 2);             // 142.6 MB
    {
        char* cbp = (char*)hcat; size_t co = 0;
        auto calloc_ = [&](size_t bytes) -> char* {
            char* p = cbp + co; co = (co + bytes + 255) & ~(size_t)255; return p;
        };
        _Float16* ycc  = (_Float16*)calloc_((size_t)Bn * 512 * 512 * 3 * 2);   // 6.3 MB
        _Float16* ic   = (_Float16*)calloc_((size_t)65536 * 512 * 2);          // 67 MB (max im2col)
        _Float16* ds0o = (_Float16*)calloc_((size_t)262144 * 32 * 2);
        _Float16* ds1o = (_Float16*)calloc_((size_t)65536 * 64 * 2);
        _Float16* ds2o = (_Float16*)calloc_((size_t)16384 * 128 * 2);
        _Float16* f256 = (_Float16*)calloc_((size_t)16384 * 256 * 2);
        _Float16* hh[3]; _Float16* hr[3];
        for (int c = 0; c < 3; ++c) hh[c] = (_Float16*)calloc_((size_t)16384 * 128 * 2);
        for (int c = 0; c < 3; ++c) hr[c] = (_Float16*)calloc_((size_t)16384 * 16 * 2);
        _Float16* stats = (_Float16*)calloc_((size_t)16384 * 32 * 2);          // total ~126 MB < 142.6 MB

        // ---- weight conversion/transpose ----------------------------------
        launch_prep(stream, ds_w[0], wt_ds[0], 48,   32,  64,   32);
        launch_prep(stream, ds_w[1], wt_ds[1], 512,  64,  512,  64);
        launch_prep(stream, ds_w[2], wt_ds[2], 1024, 128, 1024, 128);
        launch_prep(stream, fin_w,   wt_fin,   1152, 256, 1152, 256);
        for (int c = 0; c < 3; ++c) launch_prep(stream, c1_w[c], wt_c1[c], 256, 128, 256, 128);
        for (int c = 0; c < 3; ++c) launch_prep(stream, ho_w[c], wt_ho[c], 128, 3,   128, 16);
        launch_prep(stream, proj_w,   wt_proj, 15,   256, 32,   256);
        launch_prep(stream, mlp_w[0], wt_m0,   810,  256, 832,  256);
        launch_prep(stream, mlp_w[1], wt_m1,   256,  256, 256,  256);
        launch_prep(stream, mlp_w[2], wt_m2,   256,  256, 256,  256);
        launch_prep(stream, mlp_w[3], wt_m3,   1066, 256, 1088, 256);
        launch_prep(stream, mo_w,     wt_mo,   256,  3,   256,  16);

        // ---- encoder ------------------------------------------------------
        int npix = Bn * 512 * 512;
        k_ycbcr<<<(npix + 255) / 256, 256, 0, stream>>>(img, ycc, npix);

        long long tot;
        tot = (long long)262144 * 64;   // ds0: 512->256, C 3->32, K=48 pad 64
        k_im2col<<<(unsigned)((tot + 255) / 256), 256, 0, stream>>>(ycc, ic, 512, 3, 256, 4, 2, 1, 48, 64, tot);
        launch_gemm(stream, ic, 64, wt_ds[0], ds_b[0], ds0o, 32, 262144, 32, 64, 32, 1);

        tot = (long long)65536 * 512;   // ds1: 256->128, C 32->64
        k_im2col<<<(unsigned)((tot + 255) / 256), 256, 0, stream>>>(ds0o, ic, 256, 32, 128, 4, 2, 1, 512, 512, tot);
        launch_gemm(stream, ic, 512, wt_ds[1], ds_b[1], ds1o, 64, 65536, 64, 512, 64, 1);

        tot = (long long)16384 * 1024;  // ds2: 128->64, C 64->128
        k_im2col<<<(unsigned)((tot + 255) / 256), 256, 0, stream>>>(ds1o, ic, 128, 64, 64, 4, 2, 1, 1024, 1024, tot);
        launch_gemm(stream, ic, 1024, wt_ds[2], ds_b[2], ds2o, 128, 16384, 128, 1024, 128, 1);

        tot = (long long)16384 * 1152;  // final 3x3 s1, C 128->256
        k_im2col<<<(unsigned)((tot + 255) / 256), 256, 0, stream>>>(ds2o, ic, 64, 128, 64, 3, 1, 1, 1152, 1152, tot);
        launch_gemm(stream, ic, 1152, wt_fin, fin_b, f256, 256, 16384, 256, 1152, 256, 1);

        // ---- heads + stats + projection -----------------------------------
        for (int c = 0; c < 3; ++c) {
            launch_gemm(stream, f256, 256, wt_c1[c], c1_b[c], hh[c], 128, 16384, 128, 256, 128, 1);
            launch_gemm(stream, hh[c], 128, wt_ho[c], ho_b[c], hr[c], 16, 16384, 3, 128, 16, 0);
        }
        k_stats<<<(16384 + 255) / 256, 256, 0, stream>>>(hr[0], hr[1], hr[2], stats, 16384);
        launch_gemm(stream, stats, 32, wt_proj, proj_b, feat, 256, 16384, 256, 32, 256, 0);
        // conv-phase scratch inside hcat is dead from here on
    }

    // ---- pyramid, posenc, sampling (fills hcat cols 256..1087) ------------
    long long tr1 = (long long)4 * 32 * 32 * 256;
    k_resize<<<(unsigned)((tr1 + 255) / 256), 256, 0, stream>>>(feat, pyr1, 64, 32, tr1);
    long long tr2 = (long long)4 * 16 * 16 * 256;
    k_resize<<<(unsigned)((tr2 + 255) / 256), 256, 0, stream>>>(feat, pyr2, 64, 16, tr2);

    k_posenc<<<(N + 255) / 256, 256, 0, stream>>>(coords, hcat, N);
    k_sample<<<(4 * N + 255) / 256, 256, 0, stream>>>(feat, 64, coords, hcat, 256 + 42,       N);
    k_sample<<<(4 * N + 255) / 256, 256, 0, stream>>>(pyr1, 32, coords, hcat, 256 + 42 + 256, N);
    k_sample<<<(4 * N + 255) / 256, 256, 0, stream>>>(pyr2, 16, coords, hcat, 256 + 42 + 512, N);

    // ---- MLP (skip-concat realized in place in hcat) ----------------------
    launch_gemm(stream, hcat + 256, 1088, wt_m0, mlp_b[0], h1,   256,  BN, 256, 832,  256, 1);
    launch_gemm(stream, h1,         256,  wt_m1, mlp_b[1], h2,   256,  BN, 256, 256,  256, 1);
    launch_gemm(stream, h2,         256,  wt_m2, mlp_b[2], hcat, 1088, BN, 256, 256,  256, 1);
    launch_gemm(stream, hcat,       1088, wt_m3, mlp_b[3], h1,   256,  BN, 256, 1088, 256, 1); // h1 reused as h4
    launch_gemm(stream, h1,         256,  wt_mo, mo_b,     raw,  16,   BN, 3,   256,  16,  2); // tanh

    k_final<<<(BN + 255) / 256, 256, 0, stream>>>(raw, (float*)d_out, BN);
}
